// TransformerEncoder_27642409517628
// MI455X (gfx1250) — matrix operand order repaired and checked by reading, not compiled, not run on hardware
//
#include <hip/hip_runtime.h>
#include <hip/hip_bf16.h>
#include <math.h>

#define S_LEN  4096
#define D_DIM  1024
#define H_NUM  16
#define HD_DIM 64
#define FF_DIM 4096

typedef __attribute__((ext_vector_type(16))) __bf16         v16bf;
typedef __attribute__((ext_vector_type(16))) unsigned short v16u;
typedef __attribute__((ext_vector_type(8)))  unsigned short v8u;
typedef __attribute__((ext_vector_type(8)))  float          v8f;

__device__ __forceinline__ unsigned short f2bf(float f) {
    unsigned int u = __builtin_bit_cast(unsigned int, f);
    u += 0x7FFFu + ((u >> 16) & 1u);           // round-to-nearest-even
    return (unsigned short)(u >> 16);
}

// Build a 16-element bf16 fragment from two 16-byte chunks.
__device__ __forceinline__ v16bf load_frag(const unsigned short* p0,
                                           const unsigned short* p1) {
    v8u lo = *(const v8u*)p0;
    v8u hi = *(const v8u*)p1;
    v16u t;
#pragma unroll
    for (int i = 0; i < 8; ++i) { t[i] = lo[i]; t[i + 8] = hi[i]; }
    return __builtin_bit_cast(v16bf, t);
}

__device__ __forceinline__ v8f wmma_bf16(v16bf a, v16bf b, v8f c) {
    // D = A(16x32) * B(32x16) + C, f32 accumulate
    return __builtin_amdgcn_wmma_f32_16x16x32_bf16(false, a, false, b,
                                                   (short)0, c, false, false);
}

// ---------------------------------------------------------------------------
// LayerNorm: one block per row, writes bf16 row for WMMA consumption.
// ---------------------------------------------------------------------------
__global__ __launch_bounds__(256) void ln_kernel(const float* __restrict__ x,
                                                 const float* __restrict__ scale,
                                                 const float* __restrict__ bias,
                                                 unsigned short* __restrict__ yb,
                                                 int D) {
    __shared__ float red_s[8], red_ss[8];
    const int row = blockIdx.x;
    const float* xr = x + (size_t)row * D;
    float s = 0.f, ss = 0.f;
    for (int i = threadIdx.x; i < D; i += 256) {
        float v = xr[i];
        s += v; ss += v * v;
    }
#pragma unroll
    for (int o = 16; o > 0; o >>= 1) {
        s  += __shfl_xor(s,  o, 32);
        ss += __shfl_xor(ss, o, 32);
    }
    if ((threadIdx.x & 31) == 0) {
        red_s[threadIdx.x >> 5] = s;
        red_ss[threadIdx.x >> 5] = ss;
    }
    __syncthreads();
    if (threadIdx.x < 32) {
        s  = (threadIdx.x < 8) ? red_s[threadIdx.x]  : 0.f;
        ss = (threadIdx.x < 8) ? red_ss[threadIdx.x] : 0.f;
#pragma unroll
        for (int o = 4; o > 0; o >>= 1) {
            s  += __shfl_xor(s,  o, 32);
            ss += __shfl_xor(ss, o, 32);
        }
        if (threadIdx.x == 0) { red_s[0] = s; red_ss[0] = ss; }
    }
    __syncthreads();
    const float mean = red_s[0] / (float)D;
    const float var  = red_ss[0] / (float)D - mean * mean;
    const float inv  = rsqrtf(var + 1e-6f);
    for (int i = threadIdx.x; i < D; i += 256) {
        float v = (xr[i] - mean) * inv * scale[i] + bias[i];
        yb[(size_t)row * D + i] = f2bf(v);
    }
}

// ---------------------------------------------------------------------------
// Transpose + f32 -> bf16 convert: in[R][C] f32 -> out[C][R] bf16.
// ---------------------------------------------------------------------------
__global__ __launch_bounds__(256) void transpose_convert_kernel(
        const float* __restrict__ in, unsigned short* __restrict__ out,
        int R, int C) {
    int idx = blockIdx.x * 256 + threadIdx.x;
    if (idx >= R * C) return;
    int r = idx / C, c = idx % C;
    out[(size_t)c * R + r] = f2bf(in[idx]);
}

// ---------------------------------------------------------------------------
// RoPE on q,k (f32 in) -> bf16 out, layout [s][h*64+c].
// ---------------------------------------------------------------------------
__global__ __launch_bounds__(256) void rope_kernel(const float* __restrict__ qf,
                                                   const float* __restrict__ kf,
                                                   unsigned short* __restrict__ qb,
                                                   unsigned short* __restrict__ kb) {
    int idx = blockIdx.x * 256 + threadIdx.x;     // S*H*32
    if (idx >= S_LEN * H_NUM * 32) return;
    int c = idx & 31, h = (idx >> 5) & (H_NUM - 1), s = idx >> 9;
    float freq = __powf(10000.f, -(float)c * (1.f / 32.f));
    float ang = (float)s * freq;
    float sn, cs;
    __sincosf(ang, &sn, &cs);
    size_t base = (size_t)s * D_DIM + h * HD_DIM + c;
    float q1 = qf[base], q2 = qf[base + 32];
    qb[base]      = f2bf(q1 * cs - q2 * sn);
    qb[base + 32] = f2bf(q2 * cs + q1 * sn);
    float k1 = kf[base], k2 = kf[base + 32];
    kb[base]      = f2bf(k1 * cs - k2 * sn);
    kb[base + 32] = f2bf(k2 * cs + k1 * sn);
}

// ---------------------------------------------------------------------------
// bf16 WMMA GEMM: C[M][N] = A[M][K] * BT[N][K]^T + bias (+res / gelu).
// 8 waves per block (2x4 wave grid), each wave a 64x32 tile (4x2 WMMA accs):
// 8 WMMAs per 6 fragment loads -> 24 B/lane/WMMA of L2 traffic.
// MODE 0: f32 out = acc+bias.  MODE 1: f32 out = acc+bias+res.
// MODE 2: bf16 out = gelu(acc+bias).
// ---------------------------------------------------------------------------
template <int MODE>
__global__ __launch_bounds__(256) void gemm_bf16_kernel(
        const unsigned short* __restrict__ A,
        const unsigned short* __restrict__ BT,
        const float* __restrict__ bias,
        const float* __restrict__ res,
        float* __restrict__ Cf,
        unsigned short* __restrict__ Cb,
        int M, int N, int K) {
    const int lane = threadIdx.x & 31;
    const int wave = threadIdx.x >> 5;
    const int half = lane >> 4;
    const int l16  = lane & 15;
    const int mBase = blockIdx.y * 128 + (wave >> 2) * 64;   // 2 waves in M
    const int nBase = blockIdx.x * 128 + (wave & 3) * 32;    // 4 waves in N
    const int kbA = half ? 8 : 0;    // A frag: e<8 -> K=kbA+e ; e>=8 -> K=kbA+16+(e-8)
    const int kbB = half ? 8 : 0;    // B frag: e<8 -> K=kbB+e ; e>=8 -> K=kbB+16+(e-8)

    v8f acc[4][2] = {};

    const unsigned short* ap[4];
#pragma unroll
    for (int i = 0; i < 4; ++i)
        ap[i] = A + (size_t)(mBase + 16 * i + l16) * K + kbA;
    const unsigned short* b0p = BT + (size_t)(nBase + l16) * K + kbB;
    const unsigned short* b1p = BT + (size_t)(nBase + 16 + l16) * K + kbB;

    for (int kk = 0; kk < K; kk += 32) {
        __builtin_prefetch(ap[0] + kk + 256, 0, 0);
        __builtin_prefetch(b0p + kk + 256, 0, 0);
        v16bf b0 = load_frag(b0p + kk, b0p + kk + 16);
        v16bf b1 = load_frag(b1p + kk, b1p + kk + 16);
#pragma unroll
        for (int i = 0; i < 4; ++i) {
            v16bf a = load_frag(ap[i] + kk, ap[i] + kk + 16);
            acc[i][0] = wmma_bf16(a, b0, acc[i][0]);
            acc[i][1] = wmma_bf16(a, b1, acc[i][1]);
        }
    }

#pragma unroll
    for (int i = 0; i < 4; ++i) {
#pragma unroll
        for (int j = 0; j < 2; ++j) {
            const int col = nBase + 16 * j + l16;
            const float bv = bias[col];
#pragma unroll
            for (int r = 0; r < 8; ++r) {
                const int row = mBase + 16 * i + r + 8 * half;
                const size_t idx = (size_t)row * N + col;
                float v = acc[i][j][r] + bv;
                if (MODE == 0) {
                    Cf[idx] = v;
                } else if (MODE == 1) {
                    Cf[idx] = v + res[idx];
                } else {
                    float g = 0.5f * v * (1.0f + erff(v * 0.70710678118f));
                    Cb[idx] = f2bf(g);
                }
            }
        }
    }
}

// ---------------------------------------------------------------------------
// Flash attention, one wave per (32-query tile, head): two 16-row Q tiles
// share every K/V fragment -> 16 WMMAs per 8 fragment loads per 32-key chunk.
// qb/kb: [s][h*64+c] bf16.  vtb: [h*64+c][s] bf16.  ob: [s][h*64+c] bf16.
// ---------------------------------------------------------------------------
__global__ __launch_bounds__(32) void attn_kernel(
        const unsigned short* __restrict__ qb,
        const unsigned short* __restrict__ kb,
        const unsigned short* __restrict__ vtb,
        unsigned short* __restrict__ ob) {
    __shared__ unsigned short pbuf[2][16 * 32];
    const int lane = threadIdx.x;
    const int half = lane >> 4;
    const int l16  = lane & 15;
    const int h  = blockIdx.y;
    const int q0 = blockIdx.x * 32;
    const int kbA = half ? 8 : 0;
    const int kbB = half ? 8 : 0;

    // Q fragments (A layout): tile t rows = q0 + 16t + l16, c-chunks [0,32),[32,64)
    v16bf qfr[2][2];
#pragma unroll
    for (int t = 0; t < 2; ++t) {
        const unsigned short* qrow =
            qb + (size_t)(q0 + 16 * t + l16) * D_DIM + h * HD_DIM;
        qfr[t][0] = load_frag(qrow + kbA,      qrow + kbA + 16);
        qfr[t][1] = load_frag(qrow + 32 + kbA, qrow + 32 + kbA + 16);
    }

    v8f oacc[2][4] = {};
    float mrow[2][8], lsum[2][8];
#pragma unroll
    for (int t = 0; t < 2; ++t)
#pragma unroll
        for (int r = 0; r < 8; ++r) { mrow[t][r] = -1e30f; lsum[t][r] = 0.f; }

    const float scl = 0.125f;  // 1/sqrt(64)

    for (int j0 = 0; j0 < S_LEN; j0 += 32) {
        // ---- K^T fragments for 32 keys (2 key subtiles x 2 c-steps) ----
        const unsigned short* k0row = kb + (size_t)(j0 + l16) * D_DIM + h * HD_DIM;
        const unsigned short* k1row = kb + (size_t)(j0 + 16 + l16) * D_DIM + h * HD_DIM;
        v16bf kf00 = load_frag(k0row + kbB,      k0row + kbB + 16);       // keys 0-15, c 0..31
        v16bf kf01 = load_frag(k0row + 32 + kbB, k0row + 32 + kbB + 16);  // keys 0-15, c 32..63
        v16bf kf10 = load_frag(k1row + kbB,      k1row + kbB + 16);       // keys 16-31
        v16bf kf11 = load_frag(k1row + 32 + kbB, k1row + 32 + kbB + 16);

        v8f sacc[2][2] = {};
#pragma unroll
        for (int t = 0; t < 2; ++t) {
            sacc[t][0] = wmma_bf16(qfr[t][0], kf00, sacc[t][0]);
            sacc[t][0] = wmma_bf16(qfr[t][1], kf01, sacc[t][0]);
            sacc[t][1] = wmma_bf16(qfr[t][0], kf10, sacc[t][1]);
            sacc[t][1] = wmma_bf16(qfr[t][1], kf11, sacc[t][1]);
        }

        // ---- online softmax over the 32-key chunk, both q tiles ----
#pragma unroll
        for (int t = 0; t < 2; ++t) {
            float alpha[8];
#pragma unroll
            for (int r = 0; r < 8; ++r) {
                float a = sacc[t][0][r] * scl, b = sacc[t][1][r] * scl;
                float mx = fmaxf(a, b);
#pragma unroll
                for (int o = 1; o <= 8; o <<= 1) mx = fmaxf(mx, __shfl_xor(mx, o, 32));
                float mn = fmaxf(mrow[t][r], mx);
                float e0 = __expf(a - mn), e1 = __expf(b - mn);
                float rs = e0 + e1;
#pragma unroll
                for (int o = 1; o <= 8; o <<= 1) rs += __shfl_xor(rs, o, 32);
                float al = __expf(mrow[t][r] - mn);
                lsum[t][r] = lsum[t][r] * al + rs;
                mrow[t][r] = mn;
                alpha[r] = al;
                const int row = r + 8 * half;
                pbuf[t][row * 32 + l16]      = f2bf(e0);
                pbuf[t][row * 32 + 16 + l16] = f2bf(e1);
            }
#pragma unroll
            for (int d = 0; d < 4; ++d)
#pragma unroll
                for (int r = 0; r < 8; ++r) oacc[t][d][r] *= alpha[r];
        }
        __syncthreads();   // single-wave block: orders ds stores vs loads

        // ---- reload P tiles as A fragments (16x32) ----
        const unsigned short* p0row = &pbuf[0][l16 * 32];
        const unsigned short* p1row = &pbuf[1][l16 * 32];
        v16bf pf0 = load_frag(p0row + kbA, p0row + kbA + 16);
        v16bf pf1 = load_frag(p1row + kbA, p1row + kbA + 16);

        // ---- V fragments (V^T layout -> contiguous key runs), shared ----
        const unsigned short* vbase =
            vtb + ((size_t)(h * HD_DIM) + l16) * S_LEN + j0 + kbB;
#pragma unroll
        for (int d = 0; d < 4; ++d) {
            const unsigned short* vp = vbase + (size_t)(16 * d) * S_LEN;
            v16bf vb = load_frag(vp, vp + 16);
            oacc[0][d] = wmma_bf16(pf0, vb, oacc[0][d]);
            oacc[1][d] = wmma_bf16(pf1, vb, oacc[1][d]);
        }
        __syncthreads();   // protect pbuf for next iteration
    }

    // ---- normalize and write O tiles (bf16, [s][h*64+c]) ----
#pragma unroll
    for (int t = 0; t < 2; ++t) {
#pragma unroll
        for (int r = 0; r < 8; ++r) {
            const float inv = 1.f / lsum[t][r];
            const int row = q0 + 16 * t + r + 8 * half;
            const size_t base = (size_t)row * D_DIM + h * HD_DIM;
            ob[base + 0  + l16] = f2bf(oacc[t][0][r] * inv);
            ob[base + 16 + l16] = f2bf(oacc[t][1][r] * inv);
            ob[base + 32 + l16] = f2bf(oacc[t][2][r] * inv);
            ob[base + 48 + l16] = f2bf(oacc[t][3][r] * inv);
        }
    }
}

// ---------------------------------------------------------------------------
extern "C" void kernel_launch(void* const* d_in, const int* in_sizes, int n_in,
                              void* d_out, int out_size, void* d_ws, size_t ws_size,
                              hipStream_t stream) {
    (void)in_sizes; (void)n_in; (void)out_size; (void)ws_size;
    const float* x    = (const float*)d_in[0];
    const float* ln1s = (const float*)d_in[1];
    const float* ln1b = (const float*)d_in[2];
    const float* ln2s = (const float*)d_in[3];
    const float* ln2b = (const float*)d_in[4];
    const float* Wq   = (const float*)d_in[5];
    const float* bq   = (const float*)d_in[6];
    const float* Wk   = (const float*)d_in[7];
    const float* bk   = (const float*)d_in[8];
    const float* Wv   = (const float*)d_in[9];
    const float* bv   = (const float*)d_in[10];
    const float* Wo   = (const float*)d_in[11];
    const float* bo   = (const float*)d_in[12];
    const float* W1   = (const float*)d_in[13];
    const float* b1   = (const float*)d_in[14];
    const float* W2   = (const float*)d_in[15];
    const float* b2   = (const float*)d_in[16];
    float* out = (float*)d_out;

    char* ws = (char*)d_ws;
    size_t off = 0;
    auto alloc = [&](size_t n) -> char* {
        char* p = ws + off;
        off += (n + 255) & ~(size_t)255;
        return p;
    };

    unsigned short* ybf  = (unsigned short*)alloc((size_t)S_LEN * D_DIM * 2);
    unsigned short* wqT  = (unsigned short*)alloc((size_t)D_DIM * D_DIM * 2);
    unsigned short* wkT  = (unsigned short*)alloc((size_t)D_DIM * D_DIM * 2);
    unsigned short* wvT  = (unsigned short*)alloc((size_t)D_DIM * D_DIM * 2);
    unsigned short* woT  = (unsigned short*)alloc((size_t)D_DIM * D_DIM * 2);
    unsigned short* w1T  = (unsigned short*)alloc((size_t)D_DIM * FF_DIM * 2);
    unsigned short* w2T  = (unsigned short*)alloc((size_t)FF_DIM * D_DIM * 2);
    float* qf = (float*)alloc((size_t)S_LEN * D_DIM * 4);
    float* kf = (float*)alloc((size_t)S_LEN * D_DIM * 4);
    float* vf = (float*)alloc((size_t)S_LEN * D_DIM * 4);
    unsigned short* qbuf = (unsigned short*)alloc((size_t)S_LEN * D_DIM * 2);
    unsigned short* kbuf = (unsigned short*)alloc((size_t)S_LEN * D_DIM * 2);
    unsigned short* vtb  = (unsigned short*)alloc((size_t)D_DIM * S_LEN * 2);
    unsigned short* obuf = (unsigned short*)alloc((size_t)S_LEN * D_DIM * 2);
    float* x1 = (float*)alloc((size_t)S_LEN * D_DIM * 4);
    unsigned short* y2b  = (unsigned short*)alloc((size_t)S_LEN * D_DIM * 2);
    // FF activation (32MB) aliases qf+kf, which are dead after rope.
    unsigned short* hb = (unsigned short*)qf;

    // 1) LN1 -> bf16
    ln_kernel<<<dim3(S_LEN), dim3(256), 0, stream>>>(x, ln1s, ln1b, ybf, D_DIM);

    // 2) weight transpose+convert
    {
        int n = D_DIM * D_DIM, g = (n + 255) / 256;
        transpose_convert_kernel<<<dim3(g), dim3(256), 0, stream>>>(Wq, wqT, D_DIM, D_DIM);
        transpose_convert_kernel<<<dim3(g), dim3(256), 0, stream>>>(Wk, wkT, D_DIM, D_DIM);
        transpose_convert_kernel<<<dim3(g), dim3(256), 0, stream>>>(Wv, wvT, D_DIM, D_DIM);
        transpose_convert_kernel<<<dim3(g), dim3(256), 0, stream>>>(Wo, woT, D_DIM, D_DIM);
        int n2 = D_DIM * FF_DIM, g2 = (n2 + 255) / 256;
        transpose_convert_kernel<<<dim3(g2), dim3(256), 0, stream>>>(W1, w1T, D_DIM, FF_DIM);
        transpose_convert_kernel<<<dim3(g2), dim3(256), 0, stream>>>(W2, w2T, FF_DIM, D_DIM);
    }

    // 3) QKV projections (bf16 WMMA, f32 out + bias)
    dim3 gQKV(D_DIM / 128, S_LEN / 128);
    gemm_bf16_kernel<0><<<gQKV, dim3(256), 0, stream>>>(ybf, wqT, bq, nullptr, qf, nullptr, S_LEN, D_DIM, D_DIM);
    gemm_bf16_kernel<0><<<gQKV, dim3(256), 0, stream>>>(ybf, wkT, bk, nullptr, kf, nullptr, S_LEN, D_DIM, D_DIM);
    gemm_bf16_kernel<0><<<gQKV, dim3(256), 0, stream>>>(ybf, wvT, bv, nullptr, vf, nullptr, S_LEN, D_DIM, D_DIM);

    // 4) RoPE q,k -> bf16; V -> transposed bf16
    rope_kernel<<<dim3(S_LEN * H_NUM * 32 / 256), dim3(256), 0, stream>>>(qf, kf, qbuf, kbuf);
    {
        int n = S_LEN * D_DIM, g = (n + 255) / 256;
        transpose_convert_kernel<<<dim3(g), dim3(256), 0, stream>>>(vf, vtb, S_LEN, D_DIM);
    }

    // 5) flash attention (32 queries per wave)
    attn_kernel<<<dim3(S_LEN / 32, H_NUM), dim3(32), 0, stream>>>(qbuf, kbuf, vtb, obuf);

    // 6) output projection + residual
    gemm_bf16_kernel<1><<<gQKV, dim3(256), 0, stream>>>(obuf, woT, bo, x, x1, nullptr, S_LEN, D_DIM, D_DIM);

    // 7) LN2 -> bf16
    ln_kernel<<<dim3(S_LEN), dim3(256), 0, stream>>>(x1, ln2s, ln2b, y2b, D_DIM);

    // 8) FFN1 + exact GELU -> bf16
    gemm_bf16_kernel<2><<<dim3(FF_DIM / 128, S_LEN / 128), dim3(256), 0, stream>>>(
        y2b, w1T, b1, nullptr, nullptr, hb, S_LEN, FF_DIM, D_DIM);

    // 9) FFN2 + bias + residual -> out (f32)
    gemm_bf16_kernel<1><<<dim3(D_DIM / 128, S_LEN / 128), dim3(256), 0, stream>>>(
        hb, w2T, b2, x1, out, nullptr, S_LEN, D_DIM, FF_DIM);
}
